// RegionProposalNetwork_3710851744343
// MI455X (gfx1250) — matrix operand order repaired and checked by reading, not compiled, not run on hardware
//
#include <hip/hip_runtime.h>
#include <hip/hip_bf16.h>
#include <stdint.h>

typedef __attribute__((ext_vector_type(16))) _Float16 v16h;
typedef __attribute__((ext_vector_type(8)))  _Float16 v8h;
typedef __attribute__((ext_vector_type(8)))  float    v8f;
typedef __attribute__((ext_vector_type(4)))  uint32_t u32x4;
typedef __attribute__((ext_vector_type(8)))  uint32_t u32x8;

// ---------------- problem constants (from reference setup_inputs) ----------------
constexpr int NB    = 4;
constexpr int CIN   = 512;
constexpr int H     = 50;
constexpr int W     = 84;
constexpr int HP    = 52;            // padded
constexpr int WP    = 86;
constexpr int COUT  = 512;
constexpr int HWPOS = H * W;         // 4200
constexpr int M_TOT = NB * HWPOS;    // 16800
constexpr int KC    = CIN * 9;       // 4608 (im2col K)
constexpr int S_ANCH = HWPOS * 9;    // 37800 anchors per image
constexpr int NPOST = 600;
constexpr float NMS_IOU_T = 0.7f;
constexpr float MIN_SZ = 16.0f;

// ---------------- workspace layout ----------------
constexpr size_t OFF_XPAD = 0;
constexpr size_t SZ_XPAD  = (size_t)NB * HP * WP * CIN;
constexpr size_t OFF_W1T  = OFF_XPAD + SZ_XPAD;
constexpr size_t SZ_W1T   = (size_t)COUT * KC;            // [cout][k]
constexpr size_t OFF_WSLT = OFF_W1T + SZ_W1T;
constexpr size_t SZ_WSLT  = (size_t)64 * CIN;             // [col][k]
constexpr size_t OFF_FEAT = OFF_WSLT + SZ_WSLT;
constexpr size_t SZ_FEAT  = (size_t)M_TOT * COUT;         // NHWC f16
constexpr size_t HALF_TOTAL = OFF_FEAT + SZ_FEAT;
constexpr size_t FBYTE_OFF  = HALF_TOTAL * 2;
constexpr size_t OFF_FG  = 0;
constexpr size_t OFF_BOX = (size_t)NB * S_ANCH;

// ---------------- d_out layout (floats, reference return order) ----------------
constexpr size_t DO_LOGS   = 0;
constexpr size_t DO_SCORES = DO_LOGS + (size_t)NB * S_ANCH * 4;
constexpr size_t DO_ROIS   = DO_SCORES + (size_t)NB * S_ANCH * 2;
constexpr size_t DO_RIDX   = DO_ROIS + (size_t)NB * NPOST * 4;
constexpr size_t DO_ANCH   = DO_RIDX + (size_t)NB * NPOST;

// ---------------- CDNA5 async / TDM helpers ----------------
__device__ __forceinline__ void async_copy16(uint32_t lds_off, const _Float16* gptr) {
  // GLOBAL_LOAD_ASYNC_TO_LDS_B128: per-lane 16B global -> LDS, tracked by ASYNCcnt
  asm volatile("global_load_async_to_lds_b128 %0, %1, off"
               :: "v"(lds_off), "v"(gptr) : "memory");
}
__device__ __forceinline__ void wait_async0() {
  asm volatile("s_wait_asynccnt 0x0" ::: "memory");
}

// TDM: load a 2D tile (32 halfs x 64 rows, row stride KC halfs) into LDS with
// 16B padding after every 64B row chunk (-> 80B LDS row pitch). One descriptor,
// issued per-wave (EXEC ignored), tracked by TENSORcnt.
__device__ __forceinline__ void tdm_load_b(uint32_t lds_addr, const _Float16* gsrc) {
  const uint64_t ga = (uint64_t)(uintptr_t)gsrc;
  u32x4 g0;
  g0[0] = 1u;                                              // count=1, user D#
  g0[1] = lds_addr;                                        // lds_addr (bytes)
  g0[2] = (uint32_t)ga;                                    // global_addr[31:0]
  g0[3] = (uint32_t)((ga >> 32) & 0x01FFFFFFu) | 0x80000000u; // addr[56:32] | type=2
  u32x8 g1;
  g1[0] = (1u << 16)        // data_size = 2 bytes
        | (1u << 20)        // pad_enable
        | (3u << 22)        // pad_interval: 16 DWORDs (64B) between pads
        | (3u << 25);       // pad_amount: 4 DWORDs (16B)
  g1[1] = (32u & 0xFFFFu) << 16;   // tensor_dim0 = 32 (lo16 in bits 63:48)
  g1[2] = (64u << 16);             // tensor_dim0 hi=0 | tensor_dim1 = 64 (lo16)
  g1[3] = (32u << 16);             // tensor_dim1 hi=0 | tile_dim0 = 32
  g1[4] = 64u;                     // tile_dim1 = 64 | tile_dim2 = 0
  g1[5] = (uint32_t)KC;            // tensor_dim0_stride = 4608 (lo32)
  g1[6] = 0u;                      // stride hi | tensor_dim1_stride lo (unused, 2D)
  g1[7] = 0u;
  asm volatile("tensor_load_to_lds %0, %1" :: "s"(g0), "s"(g1) : "memory");
}

// ================= prep kernels =================

__global__ void k_pad_x(const float* __restrict__ x, _Float16* __restrict__ xpad) {
  size_t id = (size_t)blockIdx.x * blockDim.x + threadIdx.x;
  if (id >= SZ_XPAD) return;
  int c  = (int)(id % CIN);
  size_t t = id / CIN;
  int xx = (int)(t % WP); t /= WP;
  int yy = (int)(t % HP);
  int img = (int)(t / HP);
  float v = 0.0f;
  int ys = yy - 1, xs = xx - 1;
  if (ys >= 0 && ys < H && xs >= 0 && xs < W)
    v = x[(((size_t)img * CIN + c) * H + ys) * W + xs];
  xpad[id] = (_Float16)v;
}

__global__ void k_w1t(const float* __restrict__ W1, _Float16* __restrict__ w1t) {
  size_t id = (size_t)blockIdx.x * blockDim.x + threadIdx.x;
  if (id >= SZ_W1T) return;
  int k    = (int)(id % KC);
  int cout = (int)(id / KC);
  int tap = k >> 9;
  int cin = k & 511;
  w1t[id] = (_Float16)W1[((size_t)cout * CIN + cin) * 9 + tap];
}

__global__ void k_wslt(const float* __restrict__ Ws, const float* __restrict__ Wl,
                       _Float16* __restrict__ wslt) {
  size_t id = (size_t)blockIdx.x * blockDim.x + threadIdx.x;
  if (id >= SZ_WSLT) return;
  int k   = (int)(id % CIN);
  int col = (int)(id / CIN);
  float v = 0.0f;
  if (col < 18)      v = Ws[(size_t)col * CIN + k];
  else if (col < 54) v = Wl[(size_t)(col - 18) * CIN + k];
  wslt[id] = (_Float16)v;
}

__global__ void k_anchor(float* __restrict__ out) {
  int id = blockIdx.x * blockDim.x + threadIdx.x;
  if (id >= S_ANCH) return;
  int pos = id / 9, a = id % 9;
  int i = a / 3, j = a % 3;
  int y = pos / W, x = pos % W;
  const float ratios[3] = {0.5f, 1.0f, 2.0f};
  const float scales[3] = {8.0f, 16.0f, 32.0f};
  float hh = 16.0f * scales[j] * __fsqrt_rn(ratios[i]);
  float ww = 16.0f * scales[j] * __fsqrt_rn(1.0f / ratios[i]);
  float sx = (float)(x * 16), sy = (float)(y * 16);
  out[(size_t)id * 4 + 0] = -0.5f * hh + sx;
  out[(size_t)id * 4 + 1] = -0.5f * ww + sy;
  out[(size_t)id * 4 + 2] =  0.5f * hh + sx;
  out[(size_t)id * 4 + 3] =  0.5f * ww + sy;
}

// ================= conv3x3 GEMM: async(A) + TDM(B) -> LDS, double-buffered WMMA =====
// tiles: M=128, N=64, Kstep=32.  LDS rows padded to 40 halfs (bank-conflict-free frags)
constexpr int LDA = 40;
constexpr int LDB = 40;

__global__ __launch_bounds__(256)
void k_conv_gemm(const _Float16* __restrict__ xpad, const _Float16* __restrict__ w1t,
                 const float* __restrict__ b1, _Float16* __restrict__ feat) {
  __shared__ __align__(16) _Float16 sA[2][128 * LDA];
  __shared__ __align__(16) _Float16 sB[2][64 * LDB];

  const int tid  = threadIdx.x;
  const int lane = tid & 31;
  const int wid  = tid >> 5;
  const int g    = lane >> 4;
  const int lcol = lane & 15;
  const int nb    = blockIdx.x * 64;
  const int mbase = blockIdx.y * 128;

  // copy roles (A): thread owns rows (tid>>2) and 64+(tid>>2), 16B chunk j=tid&3
  const int jchunk = tid & 3;
  const int ar0 = tid >> 2;
  const int ar1 = 64 + (tid >> 2);

  int m0 = mbase + ar0; if (m0 > M_TOT - 1) m0 = M_TOT - 1;
  int m1 = mbase + ar1; if (m1 > M_TOT - 1) m1 = M_TOT - 1;
  const int i0 = m0 / HWPOS, p0 = m0 % HWPOS;
  const int i1 = m1 / HWPOS, p1 = m1 % HWPOS;
  const _Float16* ga0 = xpad + (((size_t)i0 * HP + p0 / W) * WP + p0 % W) * CIN;
  const _Float16* ga1 = xpad + (((size_t)i1 * HP + p1 / W) * WP + p1 % W) * CIN;
  const _Float16* gBbase = w1t + (size_t)nb * KC;   // block B panel base

  const uint32_t da0 = (uint32_t)(uintptr_t)&sA[0][ar0 * LDA + jchunk * 8];
  const uint32_t da1 = (uint32_t)(uintptr_t)&sA[0][ar1 * LDA + jchunk * 8];
  const uint32_t sb0 = (uint32_t)(uintptr_t)&sB[0][0];
  const uint32_t stepA = (uint32_t)((uintptr_t)&sA[1][0] - (uintptr_t)&sA[0][0]);
  const uint32_t stepB = (uint32_t)((uintptr_t)&sB[1][0] - (uintptr_t)&sB[0][0]);

  v8f acc[4];
#pragma unroll
  for (int f = 0; f < 4; ++f)
#pragma unroll
    for (int i = 0; i < 8; ++i) acc[f][i] = 0.0f;

  // prologue: stage K-step 0 into buffer 0
  {
    const int cin0 = jchunk * 8;
    async_copy16(da0, ga0 + cin0);
    async_copy16(da1, ga1 + cin0);
    if (wid == 0) tdm_load_b(sb0, gBbase);
  }

  int buf = 0;
  for (int kb = 0; kb < KC; kb += 32) {
    wait_async0();                          // A batch for `buf` landed (this wave)
    __builtin_amdgcn_s_wait_tensorcnt(0);   // B tile for `buf` landed (wave 0)
    __syncthreads();                        // -> globally visible

    if (kb + 32 < KC) {                     // stage next K-step into other buffer
      const int kn   = kb + 32;
      const int tap  = kn >> 9;
      const int cin0 = (kn & 511) + jchunk * 8;
      const int ty = tap / 3, tx = tap - ty * 3;
      const size_t toff = ((size_t)ty * WP + tx) * CIN + cin0;
      const uint32_t bo = (uint32_t)(buf ^ 1);
      async_copy16(da0 + bo * stepA, ga0 + toff);
      async_copy16(da1 + bo * stepA, ga1 + toff);
      if (wid == 0) tdm_load_b(sb0 + bo * stepB, gBbase + kn);
    }

    // ---- fragments from LDS (batched), 4 back-to-back WMMAs
    const _Float16* la = &sA[buf][(wid * 16 + lcol) * LDA];
    v16h afrag;
    {
      v8h c0 = *(const v8h*)(la + 8 * g);
      v8h c1 = *(const v8h*)(la + 16 + 8 * g);
#pragma unroll
      for (int i = 0; i < 8; ++i) { afrag[i] = c0[i]; afrag[i + 8] = c1[i]; }
    }
    v16h bfr[4];
#pragma unroll
    for (int f = 0; f < 4; ++f) {
      const _Float16* lb = &sB[buf][(16 * f + lcol) * LDB + 8 * g];
      v8h d0 = *(const v8h*)(lb);
      v8h d1 = *(const v8h*)(lb + 16);
#pragma unroll
      for (int i = 0; i < 8; ++i) { bfr[f][i] = d0[i]; bfr[f][i + 8] = d1[i]; }
    }
#pragma unroll
    for (int f = 0; f < 4; ++f)
      acc[f] = __builtin_amdgcn_wmma_f32_16x16x32_f16(false, afrag, false, bfr[f],
                                                      (short)0, acc[f], false, false);
    buf ^= 1;
  }

  float bias[4];
#pragma unroll
  for (int f = 0; f < 4; ++f) bias[f] = b1[nb + 16 * f + lcol];

  const int mwave = mbase + wid * 16;
#pragma unroll
  for (int r = 0; r < 8; ++r) {
    const int m = mwave + r + 8 * g;
    if (m < M_TOT) {
#pragma unroll
      for (int f = 0; f < 4; ++f) {
        const int cout = nb + 16 * f + lcol;
        float v = acc[f][r] + bias[f];
        v = fmaxf(v, 0.0f);
        feat[(size_t)m * COUT + cout] = (_Float16)v;
      }
    }
  }
}

// ================= fused 1x1 head GEMM (direct loads, small K) =================
__global__ __launch_bounds__(256)
void k_head_gemm(const _Float16* __restrict__ feat, const _Float16* __restrict__ wslt,
                 const float* __restrict__ bs, const float* __restrict__ bl,
                 float* __restrict__ o_logs, float* __restrict__ o_scores) {
  const int lane = threadIdx.x & 31;
  const int wid  = threadIdx.x >> 5;
  const int g    = lane >> 4;
  const int lcol = lane & 15;
  const int mbase = blockIdx.x * 128 + wid * 16;

  int mrow = mbase + lcol;
  if (mrow > M_TOT - 1) mrow = M_TOT - 1;
  const _Float16* arow = feat + (size_t)mrow * COUT;

  v8f acc[4];
#pragma unroll
  for (int f = 0; f < 4; ++f)
#pragma unroll
    for (int i = 0; i < 8; ++i) acc[f][i] = 0.0f;

  for (int kb = 0; kb < CIN; kb += 32) {
    v16h afrag;
    const v8h c0 = *(const v8h*)(arow + kb + 16 * g);
    const v8h c1 = *(const v8h*)(arow + kb + 8 + 16 * g);
#pragma unroll
    for (int i = 0; i < 8; ++i) { afrag[i] = c0[i]; afrag[i + 8] = c1[i]; }

    const _Float16* bp = wslt + (size_t)lcol * CIN + kb + 16 * g;
    v16h bf0 = *(const v16h*)(bp);
    v16h bf1 = *(const v16h*)(bp + (size_t)16 * CIN);
    v16h bf2 = *(const v16h*)(bp + (size_t)32 * CIN);
    v16h bf3 = *(const v16h*)(bp + (size_t)48 * CIN);
    __builtin_prefetch(arow + kb + 32, 0, 0);

    acc[0] = __builtin_amdgcn_wmma_f32_16x16x32_f16(false, afrag, false, bf0, (short)0, acc[0], false, false);
    acc[1] = __builtin_amdgcn_wmma_f32_16x16x32_f16(false, afrag, false, bf1, (short)0, acc[1], false, false);
    acc[2] = __builtin_amdgcn_wmma_f32_16x16x32_f16(false, afrag, false, bf2, (short)0, acc[2], false, false);
    acc[3] = __builtin_amdgcn_wmma_f32_16x16x32_f16(false, afrag, false, bf3, (short)0, acc[3], false, false);
  }

#pragma unroll
  for (int r = 0; r < 8; ++r) {
    const int m = mbase + r + 8 * g;
    if (m < M_TOT) {
      const int img = m / HWPOS;
      const int pos = m % HWPOS;
#pragma unroll
      for (int f = 0; f < 4; ++f) {
        const int col = 16 * f + lcol;
        if (col < 18) {
          float v = acc[f][r] + bs[col];
          o_scores[(size_t)img * HWPOS * 18 + (size_t)pos * 18 + col] = v;
        } else if (col < 54) {
          const int lc = col - 18;
          float v = acc[f][r] + bl[lc];
          o_logs[(size_t)img * HWPOS * 36 + (size_t)pos * 36 + lc] = v;
        }
      }
    }
  }
}

// ================= proposal kernels =================

__global__ void k_decode(const float* __restrict__ logs, const float* __restrict__ scoresraw,
                         const float* __restrict__ anch, const int* __restrict__ ph,
                         const int* __restrict__ pw,
                         float* __restrict__ boxes, float* __restrict__ sc) {
  int id = blockIdx.x * blockDim.x + threadIdx.x;
  if (id >= NB * S_ANCH) return;
  int s = id % S_ANCH;
  const float fh = (float)ph[0];
  const float fw = (float)pw[0];
  float a0 = anch[(size_t)s * 4 + 0], a1 = anch[(size_t)s * 4 + 1];
  float a2 = anch[(size_t)s * 4 + 2], a3 = anch[(size_t)s * 4 + 3];
  const float* lp = logs + (size_t)id * 4;
  float l0 = lp[0], l1 = lp[1], l2 = lp[2], l3 = lp[3];
  float w = a2 - a0, h = a3 - a1;
  float cx = a0 + 0.5f * w, cy = a1 + 0.5f * h;
  float ctx = l0 * w + cx, cty = l1 * h + cy;
  float nw = __expf(l2) * w, nh = __expf(l3) * h;
  float x1 = ctx - 0.5f * nw, y1 = cty - 0.5f * nh;
  float x2 = ctx + 0.5f * nw, y2 = cty + 0.5f * nh;
  x1 = fminf(fmaxf(x1, 0.0f), fw); y1 = fminf(fmaxf(y1, 0.0f), fh);
  x2 = fminf(fmaxf(x2, 0.0f), fw); y2 = fminf(fmaxf(y2, 0.0f), fh);
  bool valid = ((x2 - x1) >= MIN_SZ) && ((y2 - y1) > MIN_SZ);
  float s0 = scoresraw[(size_t)id * 2 + 0];
  float s1 = scoresraw[(size_t)id * 2 + 1];
  float fg = 1.0f / (1.0f + __expf(s0 - s1));
  boxes[(size_t)id * 4 + 0] = x1;
  boxes[(size_t)id * 4 + 1] = y1;
  boxes[(size_t)id * 4 + 2] = x2;
  boxes[(size_t)id * 4 + 3] = y2;
  sc[id] = valid ? fg : -__builtin_inff();
}

__global__ __launch_bounds__(1024)
void k_nms(const float* __restrict__ boxes_all, float* __restrict__ scores_all,
           float* __restrict__ rois, float* __restrict__ ridx) {
  const int img = blockIdx.x;
  const float* boxes = boxes_all + (size_t)img * S_ANCH * 4;
  float* scores = scores_all + (size_t)img * S_ANCH;
  const int tid = threadIdx.x;
  __shared__ float sv[1024];
  __shared__ int   si[1024];
  __shared__ float sbox[4];
  __shared__ float sarea;

  for (int it = 0; it < NPOST; ++it) {
    float bv = scores[tid];
    int   bi = tid;
    for (int s2 = tid + 1024; s2 < S_ANCH; s2 += 1024) {
      float v = scores[s2];
      if (v > bv) { bv = v; bi = s2; }
    }
    sv[tid] = bv; si[tid] = bi;
    __syncthreads();
    for (int off = 512; off > 0; off >>= 1) {
      if (tid < off) {
        float v2 = sv[tid + off]; int i2 = si[tid + off];
        if (v2 > sv[tid] || (v2 == sv[tid] && i2 < si[tid])) { sv[tid] = v2; si[tid] = i2; }
      }
      __syncthreads();
    }
    if (tid == 0) {
      const int j = si[0];
      const float v = sv[0];
      const int kv = (v != -__builtin_inff()) ? 1 : 0;
      float b0 = boxes[(size_t)j * 4 + 0], b1 = boxes[(size_t)j * 4 + 1];
      float b2 = boxes[(size_t)j * 4 + 2], b3 = boxes[(size_t)j * 4 + 3];
      sbox[0] = b0; sbox[1] = b1; sbox[2] = b2; sbox[3] = b3;
      sarea = (b2 - b0) * (b3 - b1);
      const size_t ro = (size_t)img * NPOST + it;
      rois[ro * 4 + 0] = kv ? b0 : 0.0f;
      rois[ro * 4 + 1] = kv ? b1 : 0.0f;
      rois[ro * 4 + 2] = kv ? b2 : 0.0f;
      rois[ro * 4 + 3] = kv ? b3 : 0.0f;
      ridx[ro] = kv ? (float)img : 0.0f;
    }
    __syncthreads();
    const float b0 = sbox[0], b1 = sbox[1], b2 = sbox[2], b3 = sbox[3], aj = sarea;
    for (int s2 = tid; s2 < S_ANCH; s2 += 1024) {
      float c0 = boxes[(size_t)s2 * 4 + 0], c1 = boxes[(size_t)s2 * 4 + 1];
      float c2 = boxes[(size_t)s2 * 4 + 2], c3 = boxes[(size_t)s2 * 4 + 3];
      float xx1 = fmaxf(b0, c0), yy1 = fmaxf(b1, c1);
      float xx2 = fminf(b2, c2), yy2 = fminf(b3, c3);
      float inter = fmaxf(xx2 - xx1, 0.0f) * fmaxf(yy2 - yy1, 0.0f);
      float iou = inter / (aj + (c2 - c0) * (c3 - c1) - inter + 1e-12f);
      if (iou > NMS_IOU_T) scores[s2] = -__builtin_inff();
    }
    __syncthreads();
  }
}

// ================= launch =================
extern "C" void kernel_launch(void* const* d_in, const int* in_sizes, int n_in,
                              void* d_out, int out_size, void* d_ws, size_t ws_size,
                              hipStream_t stream) {
  const float* x  = (const float*)d_in[0];
  const float* W1 = (const float*)d_in[1];
  const float* b1 = (const float*)d_in[2];
  const float* Ws = (const float*)d_in[3];
  const float* bs = (const float*)d_in[4];
  const float* Wl = (const float*)d_in[5];
  const float* bl = (const float*)d_in[6];
  const int* ph = (const int*)d_in[7];
  const int* pw = (const int*)d_in[8];

  float* out = (float*)d_out;
  _Float16* hws = (_Float16*)d_ws;
  float* fws = (float*)((char*)d_ws + FBYTE_OFF);

  _Float16* xpad = hws + OFF_XPAD;
  _Float16* w1t  = hws + OFF_W1T;
  _Float16* wslt = hws + OFF_WSLT;
  _Float16* feat = hws + OFF_FEAT;
  float* fg    = fws + OFF_FG;
  float* boxes = fws + OFF_BOX;

  float* o_logs   = out + DO_LOGS;
  float* o_scores = out + DO_SCORES;
  float* o_rois   = out + DO_ROIS;
  float* o_ridx   = out + DO_RIDX;
  float* o_anch   = out + DO_ANCH;

  k_pad_x <<<(unsigned)((SZ_XPAD + 255) / 256), 256, 0, stream>>>(x, xpad);
  k_w1t   <<<(unsigned)((SZ_W1T + 255) / 256), 256, 0, stream>>>(W1, w1t);
  k_wslt  <<<(unsigned)((SZ_WSLT + 255) / 256), 256, 0, stream>>>(Ws, Wl, wslt);
  k_anchor<<<(S_ANCH + 255) / 256, 256, 0, stream>>>(o_anch);

  k_conv_gemm<<<dim3(8, 132), 256, 0, stream>>>(xpad, w1t, b1, feat);
  k_head_gemm<<<dim3(132), 256, 0, stream>>>(feat, wslt, bs, bl, o_logs, o_scores);

  k_decode<<<(NB * S_ANCH + 255) / 256, 256, 0, stream>>>(o_logs, o_scores, o_anch, ph, pw, boxes, fg);
  k_nms<<<NB, 1024, 0, stream>>>(boxes, fg, o_rois, o_ridx);
}
